// RGCN_54176717471787
// MI455X (gfx1250) — matrix operand order repaired and checked by reading, not compiled, not run on hardware
//
#include <hip/hip_runtime.h>
#include <hip/hip_bf16.h>

// ---------------- problem constants (match reference) ----------------
#define NN   50000
#define EE   800000
#define INC  128
#define HIDC 256
#define GG   64
#define RR   3
#define METAC 38
#define BN_EPS 1e-5f

// ---------------- WMMA vector types ----------------
typedef __attribute__((ext_vector_type(16))) __bf16 v16bf;
typedef __attribute__((ext_vector_type(8)))  __bf16 v8bf;
typedef __attribute__((ext_vector_type(8)))  float  v8f;

__device__ __forceinline__ __bf16 f2bf(float f) {
    union { float f; unsigned u; } a; a.f = f;
    unsigned r = a.u + 0x7FFFu + ((a.u >> 16) & 1u);   // round-to-nearest-even
    unsigned short hi = (unsigned short)(r >> 16);
    __bf16 b;
    __builtin_memcpy(&b, &hi, 2);
    return b;
}

// ---------------------------------------------------------------------
// Convert fp32 rows [M x K] -> bf16 rows [M x KP] (KP mult of 32, zero pad),
// optionally scaling row m by 1/max(cnt[m],1)  (folds RGCN mean divide).
// ---------------------------------------------------------------------
__global__ void rows_to_bf16_kernel(const float* __restrict__ src,
                                    const float* __restrict__ cnt,
                                    __bf16* __restrict__ dst,
                                    int M, int K, int KP)
{
    const int chunks = KP >> 3;
    long long tid = (long long)blockIdx.x * blockDim.x + threadIdx.x;
    if (tid >= (long long)M * chunks) return;
    const int m = (int)(tid / chunks);
    const int c = (int)(tid % chunks) << 3;
    const float scale = cnt ? (1.0f / fmaxf(cnt[m], 1.0f)) : 1.0f;
    union { __bf16 b[8]; uint4 u; } o;
    #pragma unroll
    for (int j = 0; j < 8; ++j) {
        int k = c + j;
        float v = (k < K) ? src[(size_t)m * K + k] * scale : 0.0f;
        o.b[j] = f2bf(v);
    }
    *(uint4*)(dst + (size_t)m * KP + c) = o.u;
}

// ---------------------------------------------------------------------
// Convert + transpose weights fp32 [K x Nc] -> bf16 [NcP x KP] (zero pad).
// ---------------------------------------------------------------------
__global__ void weights_to_bf16t_kernel(const float* __restrict__ W,
                                        __bf16* __restrict__ dst,
                                        int K, int Nc, int KP, int NcP)
{
    const int chunks = KP >> 3;
    int tid = blockIdx.x * blockDim.x + threadIdx.x;
    if (tid >= NcP * chunks) return;
    const int n = tid / chunks;
    const int c = (tid % chunks) << 3;
    union { __bf16 b[8]; uint4 u; } o;
    #pragma unroll
    for (int j = 0; j < 8; ++j) {
        int k = c + j;
        float v = (k < K && n < Nc) ? W[(size_t)k * Nc + n] : 0.0f;
        o.b[j] = f2bf(v);
    }
    *(uint4*)(dst + (size_t)n * KP + c) = o.u;
}

// ---------------------------------------------------------------------
// bf16 WMMA GEMM:  C[M,Nc] (+)= A[M,KP] @ Bt[NcP,KP]^T (+ bias)
//   A row-major bf16 (K pre-padded/zeroed), Bt col-major bf16.
//   One wave per 32x64 macro-tile: 2 A frags x 4 B frags -> 8 WMMA / K-step.
//   All fragment loads unconditional (addresses clamped; garbage rows/cols
//   are never stored). EXEC all-ones at every WMMA (wave-uniform early exit).
// ---------------------------------------------------------------------
__global__ __launch_bounds__(256)
void wmma_gemm_bf16_kernel(const __bf16* __restrict__ A,
                           const __bf16* __restrict__ Bt,
                           const float*  __restrict__ bias,
                           float* __restrict__ C,
                           int M, int Nc, int KP, int NcP, int accumulate)
{
    const int lane   = threadIdx.x & 31;
    const int wave   = threadIdx.x >> 5;
    const int tilesN = (Nc + 63) >> 6;
    const int tilesM = (M + 31) >> 5;
    const int tileId = blockIdx.x * (blockDim.x >> 5) + wave;   // uniform per wave
    if (tileId >= tilesM * tilesN) return;                      // whole-wave exit

    const int tm   = tileId / tilesN;
    const int tn   = tileId % tilesN;
    const int row0 = tm << 5;
    const int col0 = tn << 6;
    const int half = lane >> 4;
    const int sub  = lane & 15;

    // clamped lane base pointers (A: 2 row subtiles, Bt: 4 col subtiles)
    const __bf16* arow[2];
    #pragma unroll
    for (int s = 0; s < 2; ++s) {
        int r = row0 + s * 16 + sub;
        if (r >= M) r = M - 1;
        arow[s] = A + (size_t)r * KP + half * 16;
    }
    const __bf16* bcol[4];
    #pragma unroll
    for (int t = 0; t < 4; ++t) {
        int n = col0 + t * 16 + sub;
        if (n >= NcP) n = NcP - 1;
        bcol[t] = Bt + (size_t)n * KP + half * 16;
    }

    v8f acc[2][4] = {};
    for (int k0 = 0; k0 < KP; k0 += 32) {
        v16bf af[2], bfrag[4];
        #pragma unroll
        for (int s = 0; s < 2; ++s) {
            union { v16bf v; v8bf h[2]; } u;
            u.h[0] = *(const v8bf*)(arow[s] + k0);        // k = k0+half*16  + 0..7
            u.h[1] = *(const v8bf*)(arow[s] + k0 + 8);    // k = k0+half*16  + 8..15
            af[s] = u.v;
        }
        #pragma unroll
        for (int t = 0; t < 4; ++t)
            bfrag[t] = *(const v16bf*)(bcol[t] + k0);     // k = k0+half*16  + 0..15
        #pragma unroll
        for (int s = 0; s < 2; ++s)
            #pragma unroll
            for (int t = 0; t < 4; ++t)
                acc[s][t] = __builtin_amdgcn_wmma_f32_16x16x32_bf16(
                    false, af[s], false, bfrag[t], (short)0, acc[s][t], false, false);
    }

    // C/D layout: VGPR r -> row = subtile_row0 + half*8 + r, col = subtile_col0 + sub
    #pragma unroll
    for (int s = 0; s < 2; ++s) {
        #pragma unroll
        for (int t = 0; t < 4; ++t) {
            const int n = col0 + t * 16 + sub;
            if (n >= Nc) continue;
            const float bv = bias ? bias[n] : 0.0f;
            #pragma unroll
            for (int r = 0; r < 8; ++r) {
                const int m = row0 + s * 16 + half * 8 + r;
                if (m < M) {
                    size_t idx = (size_t)m * Nc + n;
                    float v = acc[s][t][r] + bv;
                    if (accumulate) v += C[idx];
                    C[idx] = v;
                }
            }
        }
    }
}

// ---------------- per-(relation,dst) edge counts ----------------
__global__ void count_kernel(const int* __restrict__ dst,
                             const int* __restrict__ etype,
                             float* __restrict__ cnt)
{
    int e = blockIdx.x * blockDim.x + threadIdx.x;
    if (e >= EE) return;
    atomicAdd(&cnt[(size_t)etype[e] * NN + dst[e]], 1.0f);
}

// ---------------- aggregate x[src] into Agg[dst] for one relation ----------------
__global__ void edge_agg_kernel(const float* __restrict__ x,
                                const int* __restrict__ src,
                                const int* __restrict__ dst,
                                const int* __restrict__ etype,
                                int rel, int din,
                                float* __restrict__ Agg)
{
    const int chunks = din >> 2;                       // din % 4 == 0 always
    long long tid = (long long)blockIdx.x * blockDim.x + threadIdx.x;
    if (tid >= (long long)EE * chunks) return;
    int e = (int)(tid / chunks);
    if (etype[e] != rel) return;
    int c = (int)(tid % chunks) << 2;
    const float4 v = *(const float4*)(x + (size_t)src[e] * din + c);
    float* o = Agg + (size_t)dst[e] * din + c;
    atomicAdd(o + 0, v.x); atomicAdd(o + 1, v.y);
    atomicAdd(o + 2, v.z); atomicAdd(o + 3, v.w);
}

// ---------------- BatchNorm: per-channel stats -> (scale, shift) ----------------
__global__ void bn_stats_kernel(const float* __restrict__ h, int M, int C,
                                const float* __restrict__ g,
                                const float* __restrict__ bb,
                                float* __restrict__ scale,
                                float* __restrict__ shift)
{
    const int c = blockIdx.x;          // one block per channel
    float s = 0.0f, q = 0.0f;
    for (int m = threadIdx.x; m < M; m += blockDim.x) {
        float v = h[(size_t)m * C + c];
        s += v; q += v * v;
    }
    __shared__ float shs[256], shq[256];
    shs[threadIdx.x] = s; shq[threadIdx.x] = q;
    __syncthreads();
    for (int off = blockDim.x >> 1; off > 0; off >>= 1) {
        if ((int)threadIdx.x < off) {
            shs[threadIdx.x] += shs[threadIdx.x + off];
            shq[threadIdx.x] += shq[threadIdx.x + off];
        }
        __syncthreads();
    }
    if (threadIdx.x == 0) {
        float mu  = shs[0] / (float)M;
        float var = shq[0] / (float)M - mu * mu;   // biased var (matches jnp.var)
        float rs  = rsqrtf(var + BN_EPS);
        float sc  = g[c] * rs;
        scale[c] = sc;
        shift[c] = bb[c] - mu * sc;
    }
}

__global__ void bn_apply_kernel(float* __restrict__ h,
                                const float* __restrict__ scale,
                                const float* __restrict__ shift,
                                int M, int C)
{
    long long tid = (long long)blockIdx.x * blockDim.x + threadIdx.x;
    if (tid >= (long long)M * C) return;
    int c = (int)(tid % C);
    h[tid] = h[tid] * scale[c] + shift[c];
}

// ---------------- global mean pool ----------------
__global__ void pool_sum_kernel(const float* __restrict__ h,
                                const int* __restrict__ batch,
                                float* __restrict__ pool,
                                float* __restrict__ gcnt, int C)
{
    long long tid = (long long)blockIdx.x * blockDim.x + threadIdx.x;
    if (tid >= (long long)NN * C) return;
    int n = (int)(tid / C);
    int c = (int)(tid % C);
    int g = batch[n];
    atomicAdd(&pool[(size_t)g * C + c], h[(size_t)n * C + c]);
    if (c == 0) atomicAdd(&gcnt[g], 1.0f);
}

__global__ void pool_div_kernel(float* __restrict__ pool,
                                const float* __restrict__ gcnt, int C)
{
    int tid = blockIdx.x * blockDim.x + threadIdx.x;
    if (tid >= GG * C) return;
    pool[tid] /= fmaxf(gcnt[tid / C], 1.0f);
}

// ---------------- head: concat(pool, meta) @ W1 + b1, @ W2 + b2 ----------------
__global__ __launch_bounds__(128)
void head_kernel(const float* __restrict__ pool,  // [G, HIDC]
                 const float* __restrict__ meta,  // [G, METAC]
                 const float* __restrict__ w1,    // [HIDC+METAC, 100]
                 const float* __restrict__ b1,
                 const float* __restrict__ w2,    // [100, 1]
                 const float* __restrict__ b2,
                 float* __restrict__ out)         // [G]
{
    const int g = blockIdx.x;
    const int t = threadIdx.x;
    __shared__ float h1[128];
    float v = 0.0f;
    if (t < 100) {
        v = b1[t];
        for (int k = 0; k < HIDC; ++k)  v += pool[(size_t)g * HIDC + k] * w1[(size_t)k * 100 + t];
        for (int k = 0; k < METAC; ++k) v += meta[(size_t)g * METAC + k] * w1[(size_t)(HIDC + k) * 100 + t];
        v *= w2[t];           // fold lin2 dot product
    }
    h1[t] = (t < 100) ? v : 0.0f;
    __syncthreads();
    for (int off = 64; off > 0; off >>= 1) {
        if (t < off) h1[t] += h1[t + off];
        __syncthreads();
    }
    if (t == 0) out[g] = h1[0] + b2[0];
}

// =====================================================================
extern "C" void kernel_launch(void* const* d_in, const int* in_sizes, int n_in,
                              void* d_out, int out_size, void* d_ws, size_t ws_size,
                              hipStream_t stream)
{
    (void)in_sizes; (void)n_in; (void)out_size; (void)ws_size;

    const float* x     = (const float*)d_in[0];
    const int*   eidx  = (const int*)  d_in[2];
    const int*   esrc  = eidx;
    const int*   edst  = eidx + EE;
    const int*   etype = (const int*)  d_in[3];
    const float* meta  = (const float*)d_in[4];
    const int*   batch = (const int*)  d_in[5];

    const float *W[4], *Rw[4], *bi[4], *ga[4], *be[4];
    for (int l = 0; l < 4; ++l) {
        int base = 6 + l * 5;
        W[l]  = (const float*)d_in[base + 0];
        Rw[l] = (const float*)d_in[base + 1];
        bi[l] = (const float*)d_in[base + 2];
        ga[l] = (const float*)d_in[base + 3];
        be[l] = (const float*)d_in[base + 4];
    }
    const float* lin1_w = (const float*)d_in[26];
    const float* lin1_b = (const float*)d_in[27];
    const float* lin2_w = (const float*)d_in[28];
    const float* lin2_b = (const float*)d_in[29];

    // ---- layer shapes & padded shapes ----
    const int din_arr[4]  = { INC, 100, HIDC, HIDC };
    const int dout_arr[4] = { 100, HIDC, HIDC, HIDC };
    const int KP_arr[4]   = { 128, 128, 256, 256 };   // round_up(din, 32)
    const int NcP_arr[4]  = { 112, 256, 256, 256 };   // round_up(dout, 16)

    // ---- workspace layout ----
    float* ws    = (float*)d_ws;
    float* cnt   = ws;                                  // RR*NN
    float* Agg   = cnt  + (size_t)RR * NN;              // NN*HIDC
    float* bufA  = Agg  + (size_t)NN * HIDC;            // NN*HIDC
    float* bufB  = bufA + (size_t)NN * HIDC;            // NN*HIDC
    float* scale = bufB + (size_t)NN * HIDC;            // HIDC
    float* shift = scale + HIDC;                        // HIDC
    float* pool  = shift + HIDC;                        // GG*HIDC
    float* gcnt  = pool + (size_t)GG * HIDC;            // GG
    __bf16* Abf  = (__bf16*)(gcnt + GG);                // NN*HIDC bf16
    __bf16* Wbf  = Abf + (size_t)NN * HIDC;             // packed transposed weights

    // per-(layer, matrix) offsets into Wbf; matrix j: 0..RR-1 relations, RR = root
    size_t woff[4][RR + 1];
    {
        size_t off = 0;
        for (int l = 0; l < 4; ++l)
            for (int j = 0; j <= RR; ++j) {
                woff[l][j] = off;
                off += (size_t)NcP_arr[l] * KP_arr[l];
            }
    }

    // ---- pre-convert all weights to transposed bf16 (zero-padded) ----
    for (int l = 0; l < 4; ++l) {
        const int din = din_arr[l], dout = dout_arr[l];
        const int KP = KP_arr[l],  NcP = NcP_arr[l];
        const int wthreads = NcP * (KP >> 3);
        for (int j = 0; j <= RR; ++j) {
            const float* srcw = (j < RR) ? (W[l] + (size_t)j * din * dout) : Rw[l];
            weights_to_bf16t_kernel<<<(wthreads + 255) / 256, 256, 0, stream>>>(
                srcw, Wbf + woff[l][j], din, dout, KP, NcP);
        }
    }

    // ---- per-(relation,dst) counts (layer-independent) ----
    hipMemsetAsync(cnt, 0, (size_t)RR * NN * sizeof(float), stream);
    count_kernel<<<(EE + 255) / 256, 256, 0, stream>>>(edst, etype, cnt);

    auto launch_gemm = [&](const __bf16* A, const __bf16* Bt, const float* bp,
                           float* Cm, int M, int Nc, int KP, int NcP, int acc) {
        int tiles  = ((M + 31) / 32) * ((Nc + 63) / 64);
        int blocks = (tiles + 7) / 8;                   // 8 waves / block
        wmma_gemm_bf16_kernel<<<blocks, 256, 0, stream>>>(A, Bt, bp, Cm, M, Nc, KP, NcP, acc);
    };
    auto launch_rows = [&](const float* src, const float* cp, int K, int KP) {
        long long total = (long long)NN * (KP >> 3);
        rows_to_bf16_kernel<<<(int)((total + 255) / 256), 256, 0, stream>>>(
            src, cp, Abf, NN, K, KP);
    };

    const float* hin = x;
    for (int l = 0; l < 4; ++l) {
        const int din = din_arr[l], dout = dout_arr[l];
        const int KP = KP_arr[l],  NcP = NcP_arr[l];
        float* hout = (l & 1) ? bufB : bufA;

        // root transform + bias:  hout = hin @ Rw[l] + b[l]
        launch_rows(hin, nullptr, din, KP);
        launch_gemm(Abf, Wbf + woff[l][RR], bi[l], hout, NN, dout, KP, NcP, 0);

        // per-relation: aggregate-then-transform (linearity of mean)
        for (int r = 0; r < RR; ++r) {
            hipMemsetAsync(Agg, 0, (size_t)NN * din * sizeof(float), stream);
            long long etotal = (long long)EE * (din >> 2);
            edge_agg_kernel<<<(int)((etotal + 255) / 256), 256, 0, stream>>>(
                hin, esrc, edst, etype, r, din, Agg);
            launch_rows(Agg, cnt + (size_t)r * NN, din, KP);   // folds /max(cnt,1)
            launch_gemm(Abf, Wbf + woff[l][r], nullptr, hout, NN, dout, KP, NcP, 1);
        }

        // BatchNorm (training-mode stats over nodes), in-place
        bn_stats_kernel<<<dout, 256, 0, stream>>>(hout, NN, dout, ga[l], be[l], scale, shift);
        long long nel = (long long)NN * dout;
        bn_apply_kernel<<<(int)((nel + 255) / 256), 256, 0, stream>>>(hout, scale, shift, NN, dout);

        hin = hout;
    }

    // ---- global mean pool over batch ----
    hipMemsetAsync(pool, 0, (size_t)GG * HIDC * sizeof(float), stream);
    hipMemsetAsync(gcnt, 0, (size_t)GG * sizeof(float), stream);
    long long pel = (long long)NN * HIDC;
    pool_sum_kernel<<<(int)((pel + 255) / 256), 256, 0, stream>>>(hin, batch, pool, gcnt, HIDC);
    pool_div_kernel<<<(GG * HIDC + 255) / 256, 256, 0, stream>>>(pool, gcnt, HIDC);

    // ---- head ----
    head_kernel<<<GG, 128, 0, stream>>>(pool, meta, lin1_w, lin1_b, lin2_w, lin2_b, (float*)d_out);
}
